// DenseFlashAttention_19765439496857
// MI455X (gfx1250) — matrix operand order repaired and checked by reading, not compiled, not run on hardware
//
#include <hip/hip_runtime.h>

// ---------------- problem constants ----------------
#define NN 10000
#define EE 160000
#define DD 128
#define HH 4
#define DM 64
#define SW (DD + 8)   // padded LDS row stride (halves): 272B -> banks col*4 mod 64, conflict-free

typedef __attribute__((ext_vector_type(16))) __bf16 v16bf;
typedef __attribute__((ext_vector_type(8)))  __bf16 v8bf;
typedef __attribute__((ext_vector_type(8)))  float  v8f;

__device__ __forceinline__ v8f wmma_bf16(v16bf a, v16bf b, v8f c) {
  return __builtin_amdgcn_wmma_f32_16x16x32_bf16(false, a, false, b, (short)0, c, false, false);
}

__device__ __forceinline__ v16bf combine16(v8bf lo, v8bf hi) {
  return __builtin_shufflevector(lo, hi, 0, 1, 2, 3, 4, 5, 6, 7, 8, 9, 10, 11, 12, 13, 14, 15);
}

__device__ __forceinline__ float softplusf(float x) {
  return x > 20.f ? x : log1pf(__expf(x));
}
__device__ __forceinline__ float sigmoidf_(float x) { return 1.f / (1.f + __expf(-x)); }
__device__ __forceinline__ float siluf(float x) { return x * sigmoidf_(x); }

// monotonic float<->uint encoding for atomicMax-based segment max
__device__ __forceinline__ unsigned encf(float f) {
  unsigned u = __float_as_uint(f);
  return (u & 0x80000000u) ? ~u : (u | 0x80000000u);
}
__device__ __forceinline__ float decf(unsigned u) {
  return (u & 0x80000000u) ? __uint_as_float(u & 0x7FFFFFFFu) : __uint_as_float(~u);
}
#define ENC_NEG_INF 0x007FFFFFu  // encf(-inf)

// ---------------- K0: f32 -> bf16 convert ----------------
__global__ void cvt_f32_bf16_kernel(const float* __restrict__ in, __bf16* __restrict__ out, int n) {
  int i = blockIdx.x * blockDim.x + threadIdx.x;
  if (i < n) out[i] = (__bf16)in[i];
}

// ---------------- K1: node projections  P[h] = x @ W[h]^T  (bf16 out) ----
// grid (N/16, 1, 3*H), block (32, 8): wave y computes output n-tile y.
__global__ void proj_kernel(const __bf16* __restrict__ xb,
                            const __bf16* __restrict__ wq,
                            const __bf16* __restrict__ wr,
                            const __bf16* __restrict__ wt,
                            __bf16* __restrict__ pq,
                            __bf16* __restrict__ pr,
                            __bf16* __restrict__ pt) {
  const int lane = threadIdx.x;
  const int m0 = blockIdx.x * 16;
  const int n0 = threadIdx.y * 16;
  const int z = blockIdx.z;
  const int h = z & (HH - 1);
  const int which = z >> 2;
  const __bf16* W = (which == 0) ? wq : (which == 1) ? wr : wt;
  __bf16* P = (which == 0) ? pq : (which == 1) ? pr : pt;
  W += (size_t)h * DD * DD;
  P += (size_t)h * NN * DD;

  const int arowi = m0 + (lane & 15);
  const int kbA = (lane < 16) ? 0 : 16;
  const int colB = n0 + (lane & 15);
  const int kbB = (lane < 16) ? 0 : 16;
  const __bf16* arow = xb + (size_t)arowi * DD;
  const __bf16* brow = W + (size_t)colB * DD;

  v8f c = {};
#pragma unroll
  for (int k0 = 0; k0 < DD; k0 += 32) {
    v8bf alo = *(const v8bf*)(arow + k0 + kbA);
    v8bf ahi = *(const v8bf*)(arow + k0 + 8 + kbA);
    v16bf a = combine16(alo, ahi);
    v16bf b = *(const v16bf*)(brow + k0 + kbB);
    c = wmma_bf16(a, b, c);
  }
  const int cn = n0 + (lane & 15);
  const int mbase = m0 + ((lane < 16) ? 0 : 8);
#pragma unroll
  for (int v = 0; v < 8; v++)
    P[(size_t)(mbase + v) * DD + cn] = (__bf16)c[v];
}

// ---------------- K2: per-edge MLPs (WMMA, LDS-staged weights) + logits ----
// grid (E/128, H), block 256: 8 waves, wave w handles edge tile blockIdx.x*128 + w*16.
// dec_w1/tmp_w1 for this head staged once into LDS (32KB, padded rows).
__global__ void __launch_bounds__(256)
edge_kernel(const __bf16* __restrict__ pq,
            const int* __restrict__ edge_index,
            const float* __restrict__ edge_len,
            const __bf16* __restrict__ dw1b, const float* __restrict__ db1,
            const float* __restrict__ dw2, const float* __restrict__ db2,
            const __bf16* __restrict__ tw1b, const float* __restrict__ tb1,
            const float* __restrict__ tw2, const float* __restrict__ tb2,
            const float* __restrict__ rscore, const float* __restrict__ tscore,
            const float* __restrict__ rd_log_scale,
            const float* __restrict__ rt_bias, const float* __restrict__ rt_weight,
            const float* __restrict__ kappa,
            float* __restrict__ rl, float* __restrict__ tl) {
  const int tid = threadIdx.x;
  const int wave = tid >> 5;
  const int lane = tid & 31;
  const int h = blockIdx.y;
  const int e0 = blockIdx.x * 128 + wave * 16;
  const int* send = edge_index;
  const int* recv = edge_index + EE;

  // ---- stage dec_w1[h], tmp_w1[h] into LDS (bank-conflict-free padded rows) ----
  __shared__ __bf16 s_dw[DM * SW];
  __shared__ __bf16 s_tw[DM * SW];
  {
    // 256 threads, 128 rows total (64 per matrix), each thread copies half a row (64 halves).
    const int mat = tid >> 7;            // 0 = dec, 1 = tmp
    const int row = (tid & 127) >> 1;    // 0..63
    const int halfsel = tid & 1;         // 0..1
    const __bf16* src = (mat ? tw1b : dw1b) + ((size_t)h * DM + row) * DD + halfsel * 64;
    __bf16* dst = (mat ? s_tw : s_dw) + row * SW + halfsel * 64;
#pragma unroll
    for (int j = 0; j < 8; j++)
      *(v8bf*)(dst + j * 8) = *(const v8bf*)(src + j * 8);
  }
  __syncthreads();

  // ---- MLP GEMMs: A = rf rows (gather by receiver), B = LDS weight tiles ----
  const int myedge = e0 + (lane & 15);
  const int arowi = recv[myedge];
  const __bf16* arow = pq + ((size_t)h * NN + arowi) * DD;
  const int kbA = (lane < 16) ? 0 : 16;
  const int colL = lane & 15;
  const int kbB = (lane < 16) ? 0 : 16;

  v8f cd[4] = {{0}, {0}, {0}, {0}};
  v8f ct[4] = {{0}, {0}, {0}, {0}};
#pragma unroll
  for (int k0 = 0; k0 < DD; k0 += 32) {
    v8bf alo = *(const v8bf*)(arow + k0 + kbA);
    v8bf ahi = *(const v8bf*)(arow + k0 + 8 + kbA);
    v16bf a = combine16(alo, ahi);
#pragma unroll
    for (int t = 0; t < 4; t++) {
      const int cbase = (t * 16 + colL) * SW + k0 + kbB;
      v16bf bd = combine16(*(const v8bf*)(s_dw + cbase), *(const v8bf*)(s_dw + cbase + 8));
      cd[t] = wmma_bf16(a, bd, cd[t]);
      v16bf bt = combine16(*(const v8bf*)(s_tw + cbase), *(const v8bf*)(s_tw + cbase + 8));
      ct[t] = wmma_bf16(a, bt, ct[t]);
    }
  }

  // ---- silu + dot(w2) reduce: lane holds col t*16+colL, rows v(+8) ----
  float decv[8], tmpv[8];
#pragma unroll
  for (int v = 0; v < 8; v++) { decv[v] = 0.f; tmpv[v] = 0.f; }
#pragma unroll
  for (int t = 0; t < 4; t++) {
    const int col = t * 16 + colL;
    const float bd1 = db1[h * DM + col], wd2 = dw2[h * DM + col];
    const float bt1 = tb1[h * DM + col], wt2 = tw2[h * DM + col];
#pragma unroll
    for (int v = 0; v < 8; v++) {
      decv[v] += siluf(cd[t][v] + bd1) * wd2;
      tmpv[v] += siluf(ct[t][v] + bt1) * wt2;
    }
  }
  // butterfly sum within each 16-lane half (cols); every lane ends with its half's sums
#pragma unroll
  for (int off = 1; off < 16; off <<= 1) {
#pragma unroll
    for (int v = 0; v < 8; v++) {
      decv[v] += __shfl_xor(decv[v], off, 32);
      tmpv[v] += __shfl_xor(tmpv[v], off, 32);
    }
  }
  // bring rows 8..15 (upper half-wave) into registers of all lanes (used by lane 0)
  float dec_hi[8], tmp_hi[8];
#pragma unroll
  for (int v = 0; v < 8; v++) {
    dec_hi[v] = __shfl(decv[v], 16, 32);
    tmp_hi[v] = __shfl(tmpv[v], 16, 32);
  }

  // ---- score dot products: lane covers d = lane*4 .. lane*4+3 ----
  float rsv[4], tsv[4];
#pragma unroll
  for (int j = 0; j < 4; j++) {
    rsv[j] = rscore[h * DD + lane * 4 + j];
    tsv[j] = tscore[h * DD + lane * 4 + j];
  }
  const float sp_rd = softplusf(rd_log_scale[h]);
  const float sp_k = softplusf(kappa[0]);
  const float rtb = rt_bias[h], rtw = rt_weight[h];

#pragma unroll
  for (int m = 0; m < 16; m++) {
    const int e = e0 + m;
    const int s = send[e], r = recv[e];
    const __bf16* qs = pq + ((size_t)h * NN + s) * DD + lane * 4;
    const __bf16* qr = pq + ((size_t)h * NN + r) * DD + lane * 4;
    float dr = 0.f, dt = 0.f;
#pragma unroll
    for (int j = 0; j < 4; j++) {
      float dq = (float)qs[j] - (float)qr[j];
      dr += dq * rsv[j];
      dt += dq * tsv[j];
    }
#pragma unroll
    for (int off = 1; off < 32; off <<= 1) {
      dr += __shfl_xor(dr, off, 32);
      dt += __shfl_xor(dt, off, 32);
    }
    if (lane == 0) {
      const float dec_m = (m < 8) ? decv[m] : dec_hi[m - 8];
      const float tmp_m = (m < 8) ? tmpv[m] : tmp_hi[m - 8];
      const float el = edge_len[e];
      float rlog = dr - (sp_rd + dec_m) * el - sp_k * el;
      float rtemp = softplusf(rtb + rtw * el + tmp_m);
      rl[(size_t)h * EE + e] = rlog / (rtemp + 1e-4f);
      tl[(size_t)h * EE + e] = dt;
    }
  }
}

// ---------------- K3: init stats + accumulator -------------------------
__global__ void init_kernel(unsigned* mr, unsigned* mt, float* denr, float* dent, float* acc) {
  int i = blockIdx.x * blockDim.x + threadIdx.x;
  if (i < HH * NN) {
    mr[i] = ENC_NEG_INF; mt[i] = ENC_NEG_INF;
    denr[i] = 0.f; dent[i] = 0.f;
  }
  if (i < NN * DD) acc[i] = 0.f;
}

// ---------------- K4: segment max (atomicMax on encoded uint) ----------
__global__ void segmax_kernel(const float* __restrict__ rl, const float* __restrict__ tl,
                              const int* __restrict__ edge_index, unsigned* mr, unsigned* mt) {
  int i = blockIdx.x * blockDim.x + threadIdx.x;
  if (i >= HH * EE) return;
  int h = i / EE;
  int e = i - h * EE;
  int r = edge_index[EE + e];
  atomicMax(mr + h * NN + r, encf(rl[i]));
  atomicMax(mt + h * NN + r, encf(tl[i]));
}

// ---------------- K5a: exp + denominator -------------------------------
__global__ void exden_kernel(float* rl, float* tl, const int* __restrict__ edge_index,
                             const unsigned* __restrict__ mr, const unsigned* __restrict__ mt,
                             float* denr, float* dent) {
  int i = blockIdx.x * blockDim.x + threadIdx.x;
  if (i >= HH * EE) return;
  int h = i / EE;
  int e = i - h * EE;
  int r = edge_index[EE + e];
  float exr = __expf(rl[i] - decf(mr[h * NN + r]));
  float ext = __expf(tl[i] - decf(mt[h * NN + r]));
  rl[i] = exr; tl[i] = ext;
  atomicAdd(denr + h * NN + r, exr);
  atomicAdd(dent + h * NN + r, ext);
}

// ---------------- K5b: weighted message scatter ------------------------
// one wave per (edge, head); lane covers 4 consecutive d.
__global__ void scatter_kernel(const float* __restrict__ exr, const float* __restrict__ ext_,
                               const float* __restrict__ denr, const float* __restrict__ dent,
                               const __bf16* __restrict__ pr, const __bf16* __restrict__ pt,
                               const int* __restrict__ edge_index, const float* __restrict__ edge_len,
                               const float* __restrict__ mix_bias, const float* __restrict__ mix_scale,
                               float* acc) {
  int gid = blockIdx.x * blockDim.x + threadIdx.x;
  int wave = gid >> 5;
  int lane = gid & 31;
  if (wave >= EE * HH) return;
  int e = wave >> 2;          // H == 4
  int h = wave & (HH - 1);
  int s = edge_index[e];
  int r = edge_index[EE + e];
  float el = edge_len[e];
  float gate = sigmoidf_(mix_bias[h] + mix_scale[h] * el);
  float wr = gate * exr[(size_t)h * EE + e] / (denr[h * NN + r] + 1e-9f) * (1.0f / HH);
  float wt = (1.f - gate) * ext_[(size_t)h * EE + e] / (dent[h * NN + r] + 1e-9f) * (1.0f / HH);

  const __bf16* rs = pr + ((size_t)h * NN + s) * DD + lane * 4;
  const __bf16* rr = pr + ((size_t)h * NN + r) * DD + lane * 4;
  const __bf16* ts = pt + ((size_t)h * NN + s) * DD + lane * 4;
  const __bf16* tr = pt + ((size_t)h * NN + r) * DD + lane * 4;
  float* ao = acc + (size_t)r * DD + lane * 4;
#pragma unroll
  for (int j = 0; j < 4; j++) {
    float msg = wr * ((float)rs[j] - (float)rr[j]) + wt * ((float)ts[j] - (float)tr[j]);
    atomicAdd(ao + j, msg);
  }
}

// ---------------- K6: out = x + acc @ Wout^T (WMMA) --------------------
// grid (N/16), block (32, 8)
__global__ void out_kernel(const __bf16* __restrict__ accb, const __bf16* __restrict__ woutb,
                           const float* __restrict__ x, float* __restrict__ out) {
  const int lane = threadIdx.x;
  const int m0 = blockIdx.x * 16;
  const int n0 = threadIdx.y * 16;

  const int arowi = m0 + (lane & 15);
  const int kbA = (lane < 16) ? 0 : 16;
  const int colB = n0 + (lane & 15);
  const int kbB = (lane < 16) ? 0 : 16;
  const __bf16* arow = accb + (size_t)arowi * DD;
  const __bf16* brow = woutb + (size_t)colB * DD;

  v8f c = {};
#pragma unroll
  for (int k0 = 0; k0 < DD; k0 += 32) {
    v8bf alo = *(const v8bf*)(arow + k0 + kbA);
    v8bf ahi = *(const v8bf*)(arow + k0 + 8 + kbA);
    v16bf a = combine16(alo, ahi);
    v16bf b = *(const v16bf*)(brow + k0 + kbB);
    c = wmma_bf16(a, b, c);
  }
  const int cn = n0 + (lane & 15);
  const int mbase = m0 + ((lane < 16) ? 0 : 8);
#pragma unroll
  for (int v = 0; v < 8; v++) {
    size_t idx = (size_t)(mbase + v) * DD + cn;
    out[idx] = x[idx] + c[v];
  }
}

// ---------------- launch ----------------
extern "C" void kernel_launch(void* const* d_in, const int* in_sizes, int n_in,
                              void* d_out, int out_size, void* d_ws, size_t ws_size,
                              hipStream_t stream) {
  const float* x            = (const float*)d_in[0];
  const int*   edge_index   = (const int*)d_in[1];
  // d_in[2] = edge_vec (unused by reference)
  const float* edge_len     = (const float*)d_in[3];
  const float* Wq           = (const float*)d_in[4];
  const float* Wr           = (const float*)d_in[5];
  const float* Wt           = (const float*)d_in[6];
  const float* rscore       = (const float*)d_in[7];
  const float* tscore       = (const float*)d_in[8];
  const float* rd_log_scale = (const float*)d_in[9];
  const float* rt_bias      = (const float*)d_in[10];
  const float* rt_weight    = (const float*)d_in[11];
  const float* mix_bias     = (const float*)d_in[12];
  const float* mix_scale    = (const float*)d_in[13];
  const float* kappa        = (const float*)d_in[14];
  const float* dec_w1       = (const float*)d_in[15];
  const float* dec_b1       = (const float*)d_in[16];
  const float* dec_w2       = (const float*)d_in[17];
  const float* dec_b2       = (const float*)d_in[18];
  const float* tmp_w1       = (const float*)d_in[19];
  const float* tmp_b1       = (const float*)d_in[20];
  const float* tmp_w2       = (const float*)d_in[21];
  const float* tmp_b2       = (const float*)d_in[22];
  const float* Wout         = (const float*)d_in[23];
  float* out = (float*)d_out;

  // workspace bump allocator (256B aligned)
  size_t off = 0;
  char* base = (char*)d_ws;
  auto alloc = [&](size_t bytes) -> char* {
    off = (off + 255) & ~(size_t)255;
    char* p = base + off;
    off += bytes;
    return p;
  };
  __bf16* xb   = (__bf16*)alloc((size_t)NN * DD * 2);
  __bf16* wqb  = (__bf16*)alloc((size_t)HH * DD * DD * 2);
  __bf16* wrb  = (__bf16*)alloc((size_t)HH * DD * DD * 2);
  __bf16* wtb  = (__bf16*)alloc((size_t)HH * DD * DD * 2);
  __bf16* dw1b = (__bf16*)alloc((size_t)HH * DM * DD * 2);
  __bf16* tw1b = (__bf16*)alloc((size_t)HH * DM * DD * 2);
  __bf16* wob  = (__bf16*)alloc((size_t)DD * DD * 2);
  __bf16* pq   = (__bf16*)alloc((size_t)HH * NN * DD * 2);
  __bf16* pr   = (__bf16*)alloc((size_t)HH * NN * DD * 2);
  __bf16* pt   = (__bf16*)alloc((size_t)HH * NN * DD * 2);
  float*  rl   = (float*)alloc((size_t)HH * EE * 4);
  float*  tl   = (float*)alloc((size_t)HH * EE * 4);
  unsigned* mr = (unsigned*)alloc((size_t)HH * NN * 4);
  unsigned* mt = (unsigned*)alloc((size_t)HH * NN * 4);
  float*  denr = (float*)alloc((size_t)HH * NN * 4);
  float*  dent = (float*)alloc((size_t)HH * NN * 4);
  float*  acc  = (float*)alloc((size_t)NN * DD * 4);
  __bf16* accb = (__bf16*)alloc((size_t)NN * DD * 2);
  (void)ws_size;

  // K0: conversions to bf16
  auto cvt = [&](const float* src, __bf16* dst, int n) {
    cvt_f32_bf16_kernel<<<(n + 255) / 256, 256, 0, stream>>>(src, dst, n);
  };
  cvt(x, xb, NN * DD);
  cvt(Wq, wqb, HH * DD * DD);
  cvt(Wr, wrb, HH * DD * DD);
  cvt(Wt, wtb, HH * DD * DD);
  cvt(dec_w1, dw1b, HH * DM * DD);
  cvt(tmp_w1, tw1b, HH * DM * DD);
  cvt(Wout, wob, DD * DD);

  // K1: projections (WMMA)
  proj_kernel<<<dim3(NN / 16, 1, 3 * HH), dim3(32, 8), 0, stream>>>(xb, wqb, wrb, wtb, pq, pr, pt);

  // K3: init stats + accumulator
  {
    int n = NN * DD;  // >= H*N
    init_kernel<<<(n + 255) / 256, 256, 0, stream>>>(mr, mt, denr, dent, acc);
  }

  // K2: edge MLP + logits (WMMA, LDS-staged weights); 8 edge tiles per block
  edge_kernel<<<dim3(EE / 128, HH), 256, 0, stream>>>(
      pq, edge_index, edge_len,
      dw1b, dec_b1, dec_w2, dec_b2,
      tw1b, tmp_b1, tmp_w2, tmp_b2,
      rscore, tscore, rd_log_scale, rt_bias, rt_weight, kappa,
      rl, tl);

  // K4: segment max
  segmax_kernel<<<(HH * EE + 255) / 256, 256, 0, stream>>>(rl, tl, edge_index, mr, mt);

  // K5a: exp + denominators
  exden_kernel<<<(HH * EE + 255) / 256, 256, 0, stream>>>(rl, tl, edge_index, mr, mt, denr, dent);

  // K5b: weighted message scatter (one wave per (edge, head))
  {
    long long waves = (long long)EE * HH;
    long long blocks = (waves * 32 + 255) / 256;
    scatter_kernel<<<(int)blocks, 256, 0, stream>>>(
        rl, tl, denr, dent, pr, pt, edge_index, edge_len, mix_bias, mix_scale, acc);
  }

  // K5c: acc -> bf16
  cvt(acc, accb, NN * DD);

  // K6: final GEMM + residual
  out_kernel<<<dim3(NN / 16), dim3(32, 8), 0, stream>>>(accb, wob, x, out);

  (void)in_sizes; (void)n_in; (void)out_size;
}
